// ObjectDetectYOLOMetric_89266600280746
// MI455X (gfx1250) — matrix operand order repaired and checked by reading, not compiled, not run on hardware
//
#include <hip/hip_runtime.h>
#include <hip/hip_bf16.h>

// ---------------------------------------------------------------------------
// YOLO detect post-process for MI455X (gfx1250, wave32).
//   feats: (64,144,80,80) (64,144,40,40) (64,144,20,20) fp32
//   out:   (64,300,6) fp32  [cx,cy,w,h,conf(0 if suppressed),label]
//
// Decode is specialized per pyramid level so the channel stride is a
// compile-time constant: all 72 strided channel loads per lane lower to
// global_load_b32 with a 24-bit immediate offset off one base address
// (max offset 143*6400*4 B = 3.66 MB < 8 MB), eliminating per-load 64-bit
// address VALU in this HBM-bound (310 MB @ 23.3 TB/s ~= 13 us) kernel.
// ---------------------------------------------------------------------------

typedef __attribute__((ext_vector_type(16))) _Float16 v16h;
typedef __attribute__((ext_vector_type(8)))  float    v8f;

#define NB        64
#define NO        144
#define A_TOTAL   8400     // 6400 + 1600 + 400
#define NDET      300
#define NMS_IOU_T 0.7f
#define CONF_T    0.001f
#define CLS_OFF   10000.0f

__device__ __forceinline__ float pick8(v8f c, int idx) {
  float r = c[0];
  r = (idx == 1) ? c[1] : r;
  r = (idx == 2) ? c[2] : r;
  r = (idx == 3) ? c[3] : r;
  r = (idx == 4) ? c[4] : r;
  r = (idx == 5) ? c[5] : r;
  r = (idx == 6) ? c[6] : r;
  r = (idx == 7) ? c[7] : r;
  return r;
}

// ---------------------------------------------------------------------------
// Kernel 1: streaming decode, one wave per 16-anchor tile of ONE level.
// Lane pair (L, L+16) splits the 16 DFL bins / 80 classes of anchor L%16.
// DFL expected value via v_wmma_f32_16x16x32_f16:
//   A[m][k] = softmax prob (K=16..31 zero-padded, per-lane ISA layout)
//   B[k][n] = k for k<16 (proj vector replicated in every column)
//   D[m][n] = dist[m] for every n.
// ---------------------------------------------------------------------------
template <int HWS, int W, int LOFF, int ISTRIDE>
__global__ __launch_bounds__(256) void decode_kernel(
    const float* __restrict__ feat,
    float* __restrict__ BOXES, float* __restrict__ CONF,
    float* __restrict__ LABEL) {
  constexpr int TPI = HWS / 16;          // tiles per image at this level
  const int lane = threadIdx.x & 31;
  const int wv   = blockIdx.x * 8 + (threadIdx.x >> 5);
  const int b    = wv / TPI;
  const int ti   = wv % TPI;
  const int r    = lane & 15;            // anchor row within tile
  const int hi   = lane >> 4;            // 0: bins 0-7 / cls 0-39, 1: 8-15 / 40-79
  const int local = ti * 16 + r;         // index within this level's H*W
  const int a     = LOFF + local;        // global anchor index

  constexpr size_t cs = (size_t)HWS;     // compile-time channel stride
  const float* base = feat + (size_t)b * NO * cs + (size_t)local;

  // B fragment: lane half h holds K=8h..8h+7 (= proj 8h..8h+7) in halves 0..7,
  // halves 8..15 (K16..31) = 0 (zero pad).
  v16h bf;
#pragma unroll
  for (int k = 0; k < 16; ++k) bf[k] = (_Float16)((k < 8) ? (float)(hi * 8 + k) : 0.0f);

  v8f acc[4];
#pragma unroll
  for (int s = 0; s < 4; ++s) {
    float e[8];
    float m = -1e30f;
#pragma unroll
    for (int k = 0; k < 8; ++k) {
      float v = base[(size_t)(s * 16 + hi * 8 + k) * cs];   // imm-offset loads
      e[k] = v;
      m = fmaxf(m, v);
    }
    m = fmaxf(m, __shfl_xor(m, 16, 32));                    // max over 16 bins
    float sum = 0.f;
#pragma unroll
    for (int k = 0; k < 8; ++k) { e[k] = __expf(e[k] - m); sum += e[k]; }
    sum += __shfl_xor(sum, 16, 32);
    const float inv = 1.0f / sum;

    // 16-bit A 16x32 layout: lane m -> K0..7 in halves 0..7,
    // lane m+16 -> K8..15 in halves 0..7; halves 8..15 (K16..31) = 0.
    v16h af;
#pragma unroll
    for (int k = 0; k < 8; ++k) af[k] = (_Float16)(e[k] * inv);
#pragma unroll
    for (int k = 8; k < 16; ++k) af[k] = (_Float16)0.0f;

    v8f c = {};
    acc[s] = __builtin_amdgcn_wmma_f32_16x16x32_f16(
        /*neg_a=*/false, af, /*neg_b=*/false, bf,
        /*c_mod=*/(short)0, c, /*reuse_a=*/false, /*reuse_b=*/false);
  }

  // D layout: lanes 0-15 VGPR v -> row v; lanes 16-31 VGPR v -> row v+8.
  // All columns identical -> route row r to its lane with one shfl_xor(16).
  const int  rl      = r & 7;
  const bool ownHalf = ((r < 8) == (lane < 16));
  float dist[4];
#pragma unroll
  for (int s = 0; s < 4; ++s) {
    float own = pick8(acc[s], rl);
    float oth = __shfl_xor(own, 16, 32);
    dist[s] = ownHalf ? own : oth;
  }

  const float gx = (float)(local % W) + 0.5f;   // W is constexpr -> magic mul
  const float gy = (float)(local / W) + 0.5f;
  const float x1 = gx - dist[0], y1 = gy - dist[1];
  const float x2 = gx + dist[2], y2 = gy + dist[3];
  constexpr float stride = (float)ISTRIDE;
  const float cx = 0.5f * (x1 + x2) * stride;
  const float cy = 0.5f * (y1 + y2) * stride;
  const float bw = (x2 - x1) * stride;
  const float bh = (y2 - y1) * stride;

  // Class max/argmax on logits (sigmoid is monotone); one sigmoid at the end.
  float bm = -1e30f;
  int   bi = 0;
  const float* cb = base + (size_t)(64 + hi * 40) * cs;
#pragma unroll
  for (int i = 0; i < 40; ++i) {                 // fully unrolled imm offsets
    float v = cb[(size_t)i * cs];
    if (v > bm) { bm = v; bi = hi * 40 + i; }
  }
  float obm = __shfl_xor(bm, 16, 32);
  int   obi = __shfl_xor(bi, 16, 32);
  if (obm > bm || (obm == bm && obi < bi)) { bm = obm; bi = obi; }
  const float conf = 1.0f / (1.0f + __expf(-bm));

  if (hi == 0) {
    const size_t o = (size_t)b * A_TOTAL + (size_t)a;
    float4 bx = make_float4(cx, cy, bw, bh);
    *(float4*)(BOXES + o * 4) = bx;
    CONF[o]  = conf;
    LABEL[o] = (float)bi;
  }
}

// ---------------------------------------------------------------------------
// Kernel 2: per-image top-300 (histogram threshold + compact + bitonic sort,
// exact stable top-k order) and greedy class-aware NMS, all in LDS.
// ---------------------------------------------------------------------------
__global__ __launch_bounds__(256) void select_nms_kernel(
    const float* __restrict__ BOXES, const float* __restrict__ CONF,
    const float* __restrict__ LABEL, float* __restrict__ out) {
  const int b   = blockIdx.x;
  const int tid = threadIdx.x;

  __shared__ unsigned           hist[1024];
  __shared__ unsigned long long key[1024];
  __shared__ int   s_count, s_bucket;
  __shared__ float sx1[NDET], sy1[NDET], sx2[NDET], sy2[NDET], sarea[NDET];
  __shared__ float sconf[NDET];
  __shared__ int   sanchor[NDET], ssup[NDET];

  const float* conf = CONF + (size_t)b * A_TOTAL;

  for (int i = tid; i < 1024; i += 256) hist[i] = 0u;
  if (tid == 0) s_count = 0;
  __syncthreads();

  // conf in (0,1): float bits are monotone -> bucket = bits >> 20 (<= 1015)
  for (int a = tid; a < A_TOTAL; a += 256)
    atomicAdd(&hist[__float_as_uint(conf[a]) >> 20], 1u);
  __syncthreads();

  if (tid == 0) {
    unsigned cum = 0; int t = 1023;
    for (; t > 0; --t) { cum += hist[t]; if (cum >= NDET) break; }
    s_bucket = t;
  }
  __syncthreads();

  const unsigned tb = (unsigned)s_bucket;
  for (int a = tid; a < A_TOTAL; a += 256) {
    unsigned bits = __float_as_uint(conf[a]);
    if ((bits >> 20) >= tb) {
      int pos = atomicAdd(&s_count, 1);
      if (pos < 1024)
        key[pos] = ((unsigned long long)bits << 32) |
                   (unsigned long long)(0xFFFFFFFFu - (unsigned)a);
    }
  }
  __syncthreads();

  int n = s_count; if (n > 1024) n = 1024;
  for (int i = tid; i < 1024; i += 256) if (i >= n) key[i] = 0ull;
  __syncthreads();

  // Descending bitonic sort: higher conf first, ties -> lower anchor index.
  for (int k = 2; k <= 1024; k <<= 1) {
    for (int j = k >> 1; j > 0; j >>= 1) {
      for (int i = tid; i < 1024; i += 256) {
        int l = i ^ j;
        if (l > i) {
          unsigned long long u = key[i], v = key[l];
          bool desc = ((i & k) == 0);
          if (desc ? (u < v) : (u > v)) { key[i] = v; key[l] = u; }
        }
      }
      __syncthreads();
    }
  }

  // Top-300 -> class-offset corner boxes in LDS.
  for (int i = tid; i < NDET; i += 256) {
    unsigned long long kk = key[i];
    float cf = __uint_as_float((unsigned)(kk >> 32));
    int   a  = (int)(0xFFFFFFFFu - (unsigned)(kk & 0xFFFFFFFFull));
    sanchor[i] = a;
    sconf[i]   = cf;
    const float* bp = BOXES + ((size_t)b * A_TOTAL + (size_t)a) * 4;
    float cx = bp[0], cy = bp[1], w = bp[2], h = bp[3];
    float off = LABEL[(size_t)b * A_TOTAL + (size_t)a] * CLS_OFF;
    float x1 = cx - 0.5f * w + off, y1 = cy - 0.5f * h + off;
    float x2 = cx + 0.5f * w + off, y2 = cy + 0.5f * h + off;
    sx1[i] = x1; sy1[i] = y1; sx2[i] = x2; sy2[i] = y2;
    sarea[i] = fmaxf(x2 - x1, 0.f) * fmaxf(y2 - y1, 0.f);
    ssup[i]  = (cf > CONF_T) ? 0 : 1;
  }
  __syncthreads();

  // Greedy NMS: serial over i, parallel over j (each j owned by one thread).
  for (int i = 0; i < NDET; ++i) {
    if (!ssup[i]) {
      float x1i = sx1[i], y1i = sy1[i], x2i = sx2[i], y2i = sy2[i], ai = sarea[i];
      for (int j = tid; j < NDET; j += 256) {
        if (j > i && !ssup[j]) {
          float iw = fminf(x2i, sx2[j]) - fmaxf(x1i, sx1[j]);
          float ih = fminf(y2i, sy2[j]) - fmaxf(y1i, sy1[j]);
          float inter = fmaxf(iw, 0.f) * fmaxf(ih, 0.f);
          float iou = inter / (ai + sarea[j] - inter + 1e-7f);
          if (iou > NMS_IOU_T) ssup[j] = 1;
        }
      }
    }
    __syncthreads();
  }

  for (int i = tid; i < NDET; i += 256) {
    int a = sanchor[i];
    const float* bp = BOXES + ((size_t)b * A_TOTAL + (size_t)a) * 4;
    float* o = out + ((size_t)b * NDET + (size_t)i) * 6;
    o[0] = bp[0]; o[1] = bp[1]; o[2] = bp[2]; o[3] = bp[3];
    o[4] = ssup[i] ? 0.0f : sconf[i];
    o[5] = LABEL[(size_t)b * A_TOTAL + (size_t)a];
  }
}

// ---------------------------------------------------------------------------
extern "C" void kernel_launch(void* const* d_in, const int* in_sizes, int n_in,
                              void* d_out, int out_size, void* d_ws, size_t ws_size,
                              hipStream_t stream) {
  const float* f0 = (const float*)d_in[0];
  const float* f1 = (const float*)d_in[1];
  const float* f2 = (const float*)d_in[2];
  float* out = (float*)d_out;

  float* ws    = (float*)d_ws;
  float* BOXES = ws;                                   // 64*8400*4 floats
  float* CONF  = ws + (size_t)NB * A_TOTAL * 4;        // 64*8400
  float* LABEL = ws + (size_t)NB * A_TOTAL * 5;        // 64*8400

  // Per-level launches; each grid is an exact multiple of 8 waves so EXEC is
  // all-ones at every WMMA (no tail masking).
  decode_kernel<6400, 80, 0,    8 ><<<dim3(NB * (6400 / 16) / 8), dim3(256), 0, stream>>>(
      f0, BOXES, CONF, LABEL);
  decode_kernel<1600, 40, 6400, 16><<<dim3(NB * (1600 / 16) / 8), dim3(256), 0, stream>>>(
      f1, BOXES, CONF, LABEL);
  decode_kernel< 400, 20, 8000, 32><<<dim3(NB * ( 400 / 16) / 8), dim3(256), 0, stream>>>(
      f2, BOXES, CONF, LABEL);
  select_nms_kernel<<<dim3(NB), dim3(256), 0, stream>>>(BOXES, CONF, LABEL, out);
}
